// GRUPolicy_82789789598329
// MI455X (gfx1250) — matrix operand order repaired and checked by reading, not compiled, not run on hardware
//
#include <hip/hip_runtime.h>
#include <math.h>

// ---------------- problem constants ----------------
#define E_DIM 100
#define H_DIM 64
#define G3H   192     // 3*H
#define T_CMD 128
#define N_CMD 2048
#define T_ENV 1024
#define LIN   128
#define KPAD  128     // E padded to 128 (4 K-chunks of 32) for bf16 WMMA
#define FEAT  256     // 4*H

typedef __attribute__((ext_vector_type(16))) __bf16 v16bf;
typedef __attribute__((ext_vector_type(8)))  float  v8f;

static __device__ __forceinline__ __bf16 tobf(float x) { return (__bf16)x; }
// fast nonlinearities: v_exp_f32 + v_rcp_f32 (no IEEE divide sequence on the
// recurrence critical path; ~1ulp rcp is plenty next to bf16 matmuls)
static __device__ __forceinline__ float sigm(float x) {
  return __builtin_amdgcn_rcpf(1.0f + __expf(-x));
}
static __device__ __forceinline__ float tanh_fast(float x) {
  // tanh(x) = 2*sigmoid(2x) - 1
  return 2.0f * __builtin_amdgcn_rcpf(1.0f + __expf(-2.0f * x)) - 1.0f;
}

// ---------------- workspace layout (float elements), total ~2.6 MB ----------------
#define WS_GI    0
#define WS_ENVH  (2 * T_ENV * G3H)
#define WS_CMDH  (WS_ENVH + 2 * H_DIM)
#define WS_LOGIT (WS_CMDH + N_CMD * LIN)

// LDS strides (halfs), padded to spread banks
#define WISTR 132   // Wi  [192][132] bf16 (n-major, k-contiguous)
#define XSTR  132   // x   [16][132]  bf16
#define HSTR  68    // h   [16][68]   bf16

// smem carve (bytes)
#define SM_WI   0
#define SM_X    (SM_WI + G3H * WISTR * 2)        // 50688
#define SM_H    (SM_X + 16 * XSTR * 2)           // +4224
#define SM_END  (SM_H + 16 * HSTR * 2)           // 57088 bytes

// ============================================================
// K0: env input-side gates  gi[d][t][o] = bi[o] + env_emb[t] . eWi[o]
// ============================================================
__global__ void k0_env_gates(const float* __restrict__ env_emb,
                             const float* __restrict__ eWi_f, const float* __restrict__ ebi_f,
                             const float* __restrict__ eWi_b, const float* __restrict__ ebi_b,
                             float* __restrict__ gi) {
  int idx = blockIdx.x * blockDim.x + threadIdx.x;
  const int per = T_ENV * G3H;
  if (idx >= 2 * per) return;
  int d = idx / per, r = idx % per;
  int t = r / G3H,  o = r % G3H;
  const float* Wi = d ? eWi_b : eWi_f;
  const float* bi = d ? ebi_b : ebi_f;
  const float* x = env_emb + t * E_DIM;
  const float* w = Wi + o * E_DIM;
  float s = bi[o];
#pragma unroll 4
  for (int k = 0; k < E_DIM; ++k) s += w[k] * x[k];
  gi[idx] = s;
}

// ============================================================
// K1: encode.  blocks 0..255: cmd BiGRU tiles (16 batch rows, one direction).
//              blocks 256,257: env GRU fwd/bwd serial scans.
// ============================================================
__global__ void __launch_bounds__(128, 1)
k1_encode(const float* __restrict__ cmd_emb,
          const int*   __restrict__ cmd_lens,
          const float* __restrict__ cWi_f, const float* __restrict__ cWh_f,
          const float* __restrict__ cbi_f, const float* __restrict__ cbh_f,
          const float* __restrict__ cWi_b, const float* __restrict__ cWh_b,
          const float* __restrict__ cbi_b, const float* __restrict__ cbh_b,
          const float* __restrict__ eWh_f, const float* __restrict__ ebh_f,
          const float* __restrict__ eWh_b, const float* __restrict__ ebh_b,
          const float* __restrict__ gi,
          float* __restrict__ env_h,
          float* __restrict__ cmd_h) {
  __shared__ __align__(16) unsigned char smem[SM_END];
  const int tid  = threadIdx.x;
  const int lane = tid & 31;
  const int wv   = tid >> 5;                 // wave 0..3

  if (blockIdx.x < 256) {
    // ---------------- cmd GRU tile ----------------
    const int dir     = blockIdx.x & 1;
    const int rowbase = (blockIdx.x >> 1) * 16;
    const float* Wi  = dir ? cWi_b : cWi_f;
    const float* Wh  = dir ? cWh_b : cWh_f;
    const float* bii = dir ? cbi_b : cbi_f;
    const float* bhh = dir ? cbh_b : cbh_f;

    __bf16* Wi_l = (__bf16*)(smem + SM_WI);
    __bf16* x_l  = (__bf16*)(smem + SM_X);
    __bf16* h_l  = (__bf16*)(smem + SM_H);

    // stage Wi (bf16, n-major, K padded to 128 with zeros)
    for (int idx = tid; idx < G3H * KPAD; idx += 128) {
      int n = idx >> 7, k = idx & 127;
      Wi_l[n * WISTR + k] = (k < E_DIM) ? tobf(Wi[n * E_DIM + k]) : tobf(0.0f);
    }
    // zero the x-tile pad columns once (k = 100..127 never change)
    for (int idx = tid; idx < 16 * (KPAD - E_DIM); idx += 128) {
      int r = idx / (KPAD - E_DIM);
      int k = E_DIM + idx % (KPAD - E_DIM);
      x_l[r * XSTR + k] = tobf(0.0f);
    }

    const int col   = 16 * wv + (lane & 15);   // hidden column 0..63 owned by this lane
    const int rhalf = (lane >> 4) << 3;        // C-layout row half (0 or 8)
    const int m     = lane & 15;               // A-layout row

    // Wh B-fragments live in registers for all 128 steps
    v16bf whf[3][2];
#pragma unroll
    for (int g = 0; g < 3; ++g)
#pragma unroll
      for (int c = 0; c < 2; ++c)
#pragma unroll
        for (int j = 0; j < 16; ++j) {
          int k = c * 32 + ((lane >> 4) << 4) + j;
          whf[g][c][j] = tobf(Wh[(g * H_DIM + col) * H_DIM + k]);
        }

    int lenv[8];
#pragma unroll
    for (int v = 0; v < 8; ++v) lenv[v] = cmd_lens[rowbase + rhalf + v];

    // biases folded into accumulator init (C tile is column-broadcast: exact)
    const float b_r  = bii[col]       + bhh[col];        // r gate (input+rec bias)
    const float b_z  = bii[64 + col]  + bhh[64 + col];   // z gate
    const float b_ni = bii[128 + col];                   // n gate, input side
    const float b_nh = bhh[128 + col];                   // n gate, recurrent side

    v8f hc = {0.f, 0.f, 0.f, 0.f, 0.f, 0.f, 0.f, 0.f};  // h state (C layout)
    v16bf ha[2];                                         // h as A-fragments (bf16)
#pragma unroll
    for (int c = 0; c < 2; ++c)
#pragma unroll
      for (int j = 0; j < 16; ++j) ha[c][j] = tobf(0.0f);

    __syncthreads();

    for (int s = 0; s < T_CMD; ++s) {
      const int t = dir ? (T_CMD - 1 - s) : s;

      // ---- stage x_t: float4 global loads -> packed 4x bf16 8B DS stores ----
      // 16 rows * 25 float4 = 400 slots
      for (int slot = tid; slot < 400; slot += 128) {
        int r = slot / 25, q = slot % 25;
        float4 v = ((const float4*)cmd_emb)[(size_t)((rowbase + r) * T_CMD + t) * 25 + q];
        union { __bf16 b[4]; unsigned long long u; } pk;
        pk.b[0] = tobf(v.x); pk.b[1] = tobf(v.y); pk.b[2] = tobf(v.z); pk.b[3] = tobf(v.w);
        *(unsigned long long*)(&x_l[r * XSTR + q * 4]) = pk.u;
      }
      if (tid < 16 && s + 1 < T_CMD) {   // warm caches for next step's rows
        int tn = dir ? (T_CMD - 2 - s) : (s + 1);
        __builtin_prefetch(&cmd_emb[((size_t)(rowbase + tid) * T_CMD + tn) * E_DIM], 0, 1);
      }

      // ---- recurrent WMMAs first (registers only: overlap x staging latency) ----
      v8f a0, a1, a2, a3;   // a0:r  a1:z  a2:n-input  a3:n-recurrent
#pragma unroll
      for (int v = 0; v < 8; ++v) { a0[v] = b_r; a1[v] = b_z; a2[v] = b_ni; a3[v] = b_nh; }
#pragma unroll
      for (int c = 0; c < 2; ++c) {
        a0 = __builtin_amdgcn_wmma_f32_16x16x32_bf16(false, ha[c], false, whf[0][c], (short)0, a0, false, false);
        a1 = __builtin_amdgcn_wmma_f32_16x16x32_bf16(false, ha[c], false, whf[1][c], (short)0, a1, false, false);
        a3 = __builtin_amdgcn_wmma_f32_16x16x32_bf16(false, ha[c], false, whf[2][c], (short)0, a3, false, false);
      }
      __syncthreads();   // x_l tile visible to all waves

      // ---- input-side WMMAs, chained into the same accumulators ----
#pragma unroll
      for (int c = 0; c < 4; ++c) {
        v16bf xa;
#pragma unroll
        for (int j = 0; j < 16; ++j) {
          int k = c * 32 + ((lane >> 4) << 4) + j;
          xa[j] = x_l[m * XSTR + k];
        }
        {
          v16bf b;
#pragma unroll
          for (int j = 0; j < 16; ++j)
            b[j] = Wi_l[col * WISTR + c * 32 + ((lane >> 4) << 4) + j];
          a0 = __builtin_amdgcn_wmma_f32_16x16x32_bf16(false, xa, false, b, (short)0, a0, false, false);
        }
        {
          v16bf b;
#pragma unroll
          for (int j = 0; j < 16; ++j)
            b[j] = Wi_l[(64 + col) * WISTR + c * 32 + ((lane >> 4) << 4) + j];
          a1 = __builtin_amdgcn_wmma_f32_16x16x32_bf16(false, xa, false, b, (short)0, a1, false, false);
        }
        {
          v16bf b;
#pragma unroll
          for (int j = 0; j < 16; ++j)
            b[j] = Wi_l[(128 + col) * WISTR + c * 32 + ((lane >> 4) << 4) + j];
          a2 = __builtin_amdgcn_wmma_f32_16x16x32_bf16(false, xa, false, b, (short)0, a2, false, false);
        }
      }

      // ---- gate math (torch GRU), masked hold at padded steps ----
#pragma unroll
      for (int v = 0; v < 8; ++v) {
        float r = sigm(a0[v]);
        float z = sigm(a1[v]);
        float n = tanh_fast(a2[v] + r * a3[v]);
        float hn = (1.0f - z) * n + z * hc[v];
        hc[v] = (t < lenv[v]) ? hn : hc[v];
      }

      // ---- exchange h via LDS, rebuild A-fragments for next step ----
#pragma unroll
      for (int v = 0; v < 8; ++v) h_l[(rhalf + v) * HSTR + col] = tobf(hc[v]);
      __syncthreads();
#pragma unroll
      for (int c = 0; c < 2; ++c)
#pragma unroll
        for (int j = 0; j < 16; ++j) {
          int k = c * 32 + ((lane >> 4) << 4) + j;
          ha[c][j] = h_l[m * HSTR + k];
        }
    }
    // final hidden: cmd_h[row][dir*64 + col]
#pragma unroll
    for (int v = 0; v < 8; ++v)
      cmd_h[(size_t)(rowbase + rhalf + v) * LIN + dir * H_DIM + col] = hc[v];

  } else {
    // ---------------- env GRU serial scan (B=1), one block per direction ----------------
    const int d = blockIdx.x - 256;
    const float* Wh  = d ? eWh_b : eWh_f;
    const float* bhh = d ? ebh_b : ebh_f;
    float* Wh_l = (float*)smem;                           // [192][64] f32
    float* rec  = (float*)(smem + G3H * H_DIM * 4);       // [192]
    float* hbuf = rec + G3H;                              // [64]
    for (int idx = tid; idx < G3H * H_DIM; idx += 128) Wh_l[idx] = Wh[idx];
    for (int idx = tid; idx < H_DIM; idx += 128) hbuf[idx] = 0.0f;
    __syncthreads();
    const float* gid = gi + (size_t)d * T_ENV * G3H;
    for (int s = 0; s < T_ENV; ++s) {
      const int t = d ? (T_ENV - 1 - s) : s;
      const float* g = gid + (size_t)t * G3H;
      for (int o = tid; o < G3H; o += 128) {
        float acc = bhh[o];
        const float* w = Wh_l + o * H_DIM;
#pragma unroll 8
        for (int k = 0; k < H_DIM; ++k) acc += w[k] * hbuf[k];
        rec[o] = acc;
      }
      __syncthreads();
      if (tid < H_DIM) {
        int j = tid;
        float r = sigm(g[j] + rec[j]);
        float z = sigm(g[64 + j] + rec[64 + j]);
        float n = tanh_fast(g[128 + j] + r * rec[128 + j]);
        hbuf[j] = (1.0f - z) * n + z * hbuf[j];   // env mask always true (lens==T)
      }
      __syncthreads();
    }
    if (tid < H_DIM) env_h[d * H_DIM + tid] = hbuf[tid];
  }
}

// ============================================================
// K2: MLP  hidden = relu([env_h | cmd_h] @ W1^T + b1);  logit = hidden @ W2^T + b2
// ============================================================
__global__ void k2_mlp(const float* __restrict__ env_h, const float* __restrict__ cmd_h,
                       const float* __restrict__ W1, const float* __restrict__ b1,
                       const float* __restrict__ W2, const float* __restrict__ b2,
                       float* __restrict__ logits) {
  __shared__ __bf16 f_l[16 * 260];
  __shared__ float  hid[16 * 132];
  const int tid = threadIdx.x, lane = tid & 31, wv = tid >> 5;
  const int rowbase = blockIdx.x * 16;

  for (int idx = tid; idx < 16 * FEAT; idx += 128) {
    int r = idx >> 8, k = idx & 255;
    float v = (k < 128) ? env_h[k] : cmd_h[(size_t)(rowbase + r) * LIN + (k - 128)];
    f_l[r * 260 + k] = tobf(v);
  }
  __syncthreads();

  const int m = lane & 15;
  const int n0 = 16 * wv + (lane & 15);
  const int n1 = n0 + 64;
  v8f acc0, acc1;
  const float bb0 = b1[n0], bb1 = b1[n1];
#pragma unroll
  for (int v = 0; v < 8; ++v) { acc0[v] = bb0; acc1[v] = bb1; }
#pragma unroll
  for (int c = 0; c < 8; ++c) {
    v16bf a;
#pragma unroll
    for (int j = 0; j < 16; ++j) {
      int k = c * 32 + ((lane >> 4) << 4) + j;
      a[j] = f_l[m * 260 + k];
    }
    v16bf b0, b1v;
#pragma unroll
    for (int j = 0; j < 16; ++j) {
      int k = c * 32 + ((lane >> 4) << 4) + j;
      b0[j]  = tobf(W1[(size_t)n0 * FEAT + k]);
      b1v[j] = tobf(W1[(size_t)n1 * FEAT + k]);
    }
    acc0 = __builtin_amdgcn_wmma_f32_16x16x32_bf16(false, a, false, b0,  (short)0, acc0, false, false);
    acc1 = __builtin_amdgcn_wmma_f32_16x16x32_bf16(false, a, false, b1v, (short)0, acc1, false, false);
  }
  const int rhalf = (lane >> 4) << 3;
#pragma unroll
  for (int v = 0; v < 8; ++v) {
    hid[(rhalf + v) * 132 + n0] = fmaxf(acc0[v], 0.0f);
    hid[(rhalf + v) * 132 + n1] = fmaxf(acc1[v], 0.0f);
  }
  __syncthreads();
  if (tid < 16) {
    float s = b2[0];
#pragma unroll 8
    for (int j = 0; j < LIN; ++j) s += hid[tid * 132 + j] * W2[j];
    logits[rowbase + tid] = s;
  }
}

// ============================================================
// K3: softmax over 2048 logits, single block
// ============================================================
__global__ void k3_softmax(const float* __restrict__ logits, float* __restrict__ out) {
  __shared__ float red[256];
  const int tid = threadIdx.x;
  float mx = -INFINITY;
  for (int i = tid; i < N_CMD; i += 256) mx = fmaxf(mx, logits[i]);
  red[tid] = mx; __syncthreads();
  for (int s = 128; s > 0; s >>= 1) { if (tid < s) red[tid] = fmaxf(red[tid], red[tid + s]); __syncthreads(); }
  const float M = red[0]; __syncthreads();
  float sm = 0.0f;
  for (int i = tid; i < N_CMD; i += 256) sm += __expf(logits[i] - M);
  red[tid] = sm; __syncthreads();
  for (int s = 128; s > 0; s >>= 1) { if (tid < s) red[tid] += red[tid + s]; __syncthreads(); }
  const float inv = 1.0f / red[0];
  for (int i = tid; i < N_CMD; i += 256) out[i] = __expf(logits[i] - M) * inv;
}

// ============================================================
extern "C" void kernel_launch(void* const* d_in, const int* in_sizes, int n_in,
                              void* d_out, int out_size, void* d_ws, size_t ws_size,
                              hipStream_t stream) {
  (void)in_sizes; (void)n_in; (void)out_size; (void)ws_size;
  const float* env_emb  = (const float*)d_in[0];
  const float* cmd_emb  = (const float*)d_in[1];
  const int*   cmd_lens = (const int*)d_in[2];
  const float* eWi_f = (const float*)d_in[3],  *eWh_f = (const float*)d_in[4];
  const float* ebi_f = (const float*)d_in[5],  *ebh_f = (const float*)d_in[6];
  const float* eWi_b = (const float*)d_in[7],  *eWh_b = (const float*)d_in[8];
  const float* ebi_b = (const float*)d_in[9],  *ebh_b = (const float*)d_in[10];
  const float* cWi_f = (const float*)d_in[11], *cWh_f = (const float*)d_in[12];
  const float* cbi_f = (const float*)d_in[13], *cbh_f = (const float*)d_in[14];
  const float* cWi_b = (const float*)d_in[15], *cWh_b = (const float*)d_in[16];
  const float* cbi_b = (const float*)d_in[17], *cbh_b = (const float*)d_in[18];
  const float* W1 = (const float*)d_in[19], *b1 = (const float*)d_in[20];
  const float* W2 = (const float*)d_in[21], *b2 = (const float*)d_in[22];

  float* ws     = (float*)d_ws;       // needs ~2.6 MB
  float* gi     = ws + WS_GI;
  float* env_h  = ws + WS_ENVH;
  float* cmd_h  = ws + WS_CMDH;
  float* logits = ws + WS_LOGIT;

  k0_env_gates<<<(2 * T_ENV * G3H + 127) / 128, 128, 0, stream>>>(
      env_emb, eWi_f, ebi_f, eWi_b, ebi_b, gi);
  k1_encode<<<258, 128, 0, stream>>>(
      cmd_emb, cmd_lens,
      cWi_f, cWh_f, cbi_f, cbh_f, cWi_b, cWh_b, cbi_b, cbh_b,
      eWh_f, ebh_f, eWh_b, ebh_b,
      gi, env_h, cmd_h);
  k2_mlp<<<N_CMD / 16, 128, 0, stream>>>(env_h, cmd_h, W1, b1, W2, b2, logits);
  k3_softmax<<<1, 256, 0, stream>>>(logits, (float*)d_out);
}
